// CustomGCN_OnlyNFeatSumMsg_16492674417025
// MI455X (gfx1250) — matrix operand order repaired and checked by reading, not compiled, not run on hardware
//
#include <hip/hip_runtime.h>

#define N_NODES  100000
#define N_EDGES  1600000
#define N_GRAPHS 64
#define IN_F     128
#define H_F      128
#define N_CLS    10

typedef __attribute__((ext_vector_type(16))) _Float16 v16h;
typedef __attribute__((ext_vector_type(8)))  _Float16 h8;
typedef __attribute__((ext_vector_type(8)))  float    v8f;
typedef __attribute__((ext_vector_type(4)))  float    f4;

// Types matching the async-to-LDS builtin signature (vV4i*1 V4i*3 Ii Ii)
typedef int v4i __attribute__((vector_size(16)));
typedef __attribute__((address_space(1))) v4i gas_v4i;   // global
typedef __attribute__((address_space(3))) v4i las_v4i;   // LDS

#define A_STRIDE 132   // f32 tile stride (528B): bank step 4, 16B-aligned
#define W_STRIDE 136   // f16 W^T stride (272B): bank step 4 (2-way), 16B-aligned

#if __has_builtin(__builtin_amdgcn_global_load_async_to_lds_b128) && \
    __has_builtin(__builtin_amdgcn_s_wait_asynccnt)
#define USE_ASYNC_LDS 1
#else
#define USE_ASYNC_LDS 0
#endif

// ---------------------------------------------------------------------------
// Edge scatter: out[dst[e]] += x[src[e]]   (one wave per edge, float4 per lane)
// ---------------------------------------------------------------------------
__global__ __launch_bounds__(256)
void gcn_scatter_add(const float* __restrict__ x, const int* __restrict__ src,
                     const int* __restrict__ dst, float* __restrict__ out, int nEdges)
{
    int e    = (int)((blockIdx.x * 256u + threadIdx.x) >> 5);
    int lane = threadIdx.x & 31;
    if (e >= nEdges) return;
    int s = src[e];
    int d = dst[e];
    float4 v = ((const float4*)(x + (size_t)s * H_F))[lane];
    float* o = out + (size_t)d * H_F + lane * 4;
    unsafeAtomicAdd(o + 0, v.x);
    unsafeAtomicAdd(o + 1, v.y);
    unsafeAtomicAdd(o + 2, v.z);
    unsafeAtomicAdd(o + 3, v.w);
}

// ---------------------------------------------------------------------------
// out = relu(A @ W + bias), A:[nRows,128] f32, W:[128,128] f32 (row-major K,N)
// Block = 256 threads = 8 waves; block tile = 128 rows x 128 cols.
// A tile staged f32 via async Global->LDS (ASYNCcnt); converted to f16 at
// fragment build. W^T staged f16 via ds stores. 32 WMMAs per wave, K=128.
// ---------------------------------------------------------------------------
__global__ __launch_bounds__(256)
void gcn_gemm_bias_relu(const float* __restrict__ A, const float* __restrict__ W,
                        const float* __restrict__ bias, float* __restrict__ out,
                        int nRows)
{
    __shared__ float    Asm[128 * A_STRIDE];   // A tile, f32, row-major [r][k]
    __shared__ _Float16 Wsm[128 * W_STRIDE];   // W^T, f16: Wsm[n][k] = W[k][n]

    const int tid   = threadIdx.x;
    const int lane  = tid & 31;
    const int wave  = tid >> 5;
    const int mBase = blockIdx.x * 128;

    // Stage W^T into LDS as f16 (coalesced global read)
    for (int idx = tid; idx < 128 * 128; idx += 256) {
        int k = idx >> 7, n = idx & 127;
        Wsm[n * W_STRIDE + k] = (_Float16)W[idx];
    }
    // Stage A tile into LDS as f32, 16B per lane per iteration
    for (int idx = tid; idx < 128 * 32; idx += 256) {
        int r   = idx >> 5;
        int c4  = idx & 31;
        int row = mBase + r;
        float* ldst = &Asm[r * A_STRIDE + c4 * 4];
        if (row < nRows) {
#if USE_ASYNC_LDS
            __builtin_amdgcn_global_load_async_to_lds_b128(
                (gas_v4i*)(A + (size_t)row * H_F + c4 * 4),
                (las_v4i*)ldst, 0, 0);
#else
            *(f4*)ldst = ((const f4*)(A + (size_t)row * H_F))[c4];
#endif
        } else {
            f4 z = {};
            *(f4*)ldst = z;
        }
    }
#if USE_ASYNC_LDS
    __builtin_amdgcn_s_wait_asynccnt(0);
#endif
    __syncthreads();

    const int nl   = lane & 15;      // N (or M-row) index within 16-wide tile
    const int hi   = lane >> 4;      // half-wave select
    const int aOff = hi * 8;         // A: lanes 16-31 hold K {8..15, 24..31}
    const int bOff = hi * 8;         // B: lanes 16-31 hold K {8..15, 24..31}
    const int aRow = wave * 16 + nl;

    v8f acc[8];
    #pragma unroll
    for (int t = 0; t < 8; ++t) acc[t] = (v8f){};

    #pragma unroll
    for (int kb = 0; kb < 128; kb += 32) {
        // A fragment: f32 LDS loads + cvt. 16 halfs = K {kb+aOff..+8} ++ {kb+16+aOff..+8}
        const float* ap = &Asm[aRow * A_STRIDE + kb + aOff];
        f4 p0 = *(const f4*)(ap + 0);
        f4 p1 = *(const f4*)(ap + 4);
        f4 p2 = *(const f4*)(ap + 16);
        f4 p3 = *(const f4*)(ap + 20);
        v16h a;
        #pragma unroll
        for (int i = 0; i < 4; ++i) {
            a[i]      = (_Float16)p0[i];
            a[4 + i]  = (_Float16)p1[i];
            a[8 + i]  = (_Float16)p2[i];
            a[12 + i] = (_Float16)p3[i];
        }

        #pragma unroll
        for (int t = 0; t < 8; ++t) {
            int n = t * 16 + nl;     // column this lane supplies for B
            h8 blo = *(const h8*)&Wsm[n * W_STRIDE + kb + bOff];
            h8 bhi = *(const h8*)&Wsm[n * W_STRIDE + kb + bOff + 16];
            v16h b;
            #pragma unroll
            for (int i = 0; i < 8; ++i) { b[i] = blo[i]; b[i + 8] = bhi[i]; }
            acc[t] = __builtin_amdgcn_wmma_f32_16x16x32_f16(
                false, a, false, b, (short)0, acc[t], false, false);
        }
    }

    // Epilogue: C layout — VGPR r: M = r + 8*hi, N = lane&15 (within 16x16 tile)
    const int outRow0 = mBase + wave * 16 + hi * 8;
    #pragma unroll
    for (int t = 0; t < 8; ++t) {
        int col = t * 16 + nl;
        float bv = bias[col];
        #pragma unroll
        for (int r = 0; r < 8; ++r) {
            int row = outRow0 + r;
            if (row < nRows) {
                float v = acc[t][r] + bv;
                out[(size_t)row * H_F + col] = v > 0.f ? v : 0.f;
            }
        }
    }
}

// ---------------------------------------------------------------------------
// Per-graph sums + counts (one wave per node, atomics into L2-resident 33KB)
// ---------------------------------------------------------------------------
__global__ __launch_bounds__(256)
void gcn_pool_sum(const float* __restrict__ h, const int* __restrict__ gid,
                  float* __restrict__ sums, float* __restrict__ cnt, int nNodes)
{
    int node = (int)((blockIdx.x * 256u + threadIdx.x) >> 5);
    int lane = threadIdx.x & 31;
    if (node >= nNodes) return;
    int g = gid[node];
    float4 v = ((const float4*)(h + (size_t)node * H_F))[lane];
    float* o = sums + g * H_F + lane * 4;
    unsafeAtomicAdd(o + 0, v.x);
    unsafeAtomicAdd(o + 1, v.y);
    unsafeAtomicAdd(o + 2, v.z);
    unsafeAtomicAdd(o + 3, v.w);
    if (lane == 0) unsafeAtomicAdd(&cnt[g], 1.0f);
}

// ---------------------------------------------------------------------------
// Head: out[g,c] = (sums[g]/max(cnt[g],1)) @ Wp + bp   (640 tiny dots)
// ---------------------------------------------------------------------------
__global__ __launch_bounds__(256)
void gcn_head(const float* __restrict__ sums, const float* __restrict__ cnt,
              const float* __restrict__ Wp, const float* __restrict__ bp,
              float* __restrict__ out)
{
    int t = blockIdx.x * 256 + threadIdx.x;
    if (t >= N_GRAPHS * N_CLS) return;
    int g = t / N_CLS, c = t % N_CLS;
    float inv = 1.0f / fmaxf(cnt[g], 1.0f);
    float a = bp[c];
    #pragma unroll 4
    for (int k = 0; k < H_F; ++k)
        a += (sums[g * H_F + k] * inv) * Wp[k * N_CLS + c];
    out[t] = a;
}

// ---------------------------------------------------------------------------
extern "C" void kernel_launch(void* const* d_in, const int* in_sizes, int n_in,
                              void* d_out, int out_size, void* d_ws, size_t ws_size,
                              hipStream_t stream)
{
    const float* feat = (const float*)d_in[0];
    const float* W1   = (const float*)d_in[1];
    const float* b1   = (const float*)d_in[2];
    const float* W2   = (const float*)d_in[3];
    const float* b2   = (const float*)d_in[4];
    const float* Wp   = (const float*)d_in[5];
    const float* bp   = (const float*)d_in[6];
    const int*   src  = (const int*)d_in[7];
    const int*   dst  = (const int*)d_in[8];
    const int*   gid  = (const int*)d_in[9];

    float* agg  = (float*)d_ws;                       // [N_NODES,128]
    float* hbuf = agg  + (size_t)N_NODES * H_F;       // [N_NODES,128]
    float* sums = hbuf + (size_t)N_NODES * H_F;       // [64,128]
    float* cnt  = sums + N_GRAPHS * H_F;              // [64]

    const size_t aggBytes = (size_t)N_NODES * H_F * sizeof(float);
    const int edgeBlocks  = (N_EDGES + 7) / 8;        // 8 edges (waves) per block
    const int gemmBlocks  = (N_NODES + 127) / 128;
    const int poolBlocks  = (N_NODES + 7) / 8;

    // Layer 1: agg = scatter_sum(feat); h = relu(agg @ W1 + b1)
    (void)hipMemsetAsync(agg, 0, aggBytes, stream);
    gcn_scatter_add<<<edgeBlocks, 256, 0, stream>>>(feat, src, dst, agg, N_EDGES);
    gcn_gemm_bias_relu<<<gemmBlocks, 256, 0, stream>>>(agg, W1, b1, hbuf, N_NODES);

    // Layer 2: agg = scatter_sum(h); h = relu(agg @ W2 + b2)
    (void)hipMemsetAsync(agg, 0, aggBytes, stream);
    gcn_scatter_add<<<edgeBlocks, 256, 0, stream>>>(hbuf, src, dst, agg, N_EDGES);
    gcn_gemm_bias_relu<<<gemmBlocks, 256, 0, stream>>>(agg, W2, b2, hbuf, N_NODES);

    // Mean pool + head
    (void)hipMemsetAsync(sums, 0, (N_GRAPHS * H_F + N_GRAPHS) * sizeof(float), stream);
    gcn_pool_sum<<<poolBlocks, 256, 0, stream>>>(hbuf, gid, sums, cnt, N_NODES);
    gcn_head<<<(N_GRAPHS * N_CLS + 255) / 256, 256, 0, stream>>>(sums, cnt, Wp, bp,
                                                                 (float*)d_out);
}
